// SaptioTemporalNN_multitime_78907139162497
// MI455X (gfx1250) — matrix operand order repaired and checked by reading, not compiled, not run on hardware
//
#include <hip/hip_runtime.h>
#include <hip/hip_bf16.h>

// Problem constants (from reference setup_inputs)
#define NXDIM  2048
#define NZDIM  32
#define NDDIM  8
#define NSTEPS 16
#define NTP1   25          // NT + 1
#define KSPLIT 8
#define KSLICE (NXDIM / KSPLIT)   // 256

typedef __attribute__((ext_vector_type(16))) __bf16 v16bf;
typedef __attribute__((ext_vector_type(8)))  __bf16 v8bf;
typedef __attribute__((ext_vector_type(8)))  float  v8f;

static __device__ __forceinline__ unsigned short f2bf(float f) {
    union { __bf16 h; unsigned short u; } cv;
    cv.h = (__bf16)f;
    return cv.u;
}

// ---------------------------------------------------------------------------
// Pass 1: gather the 16 needed time slices of `relations` once, convert to
// bf16, store time-major row-major: relT[t][x*2048 + y].
// relations index: ((x*NR_REL + 0)*NX + y)*NTP1 + (start_time + t)
// ---------------------------------------------------------------------------
__global__ void __launch_bounds__(256)
transpose_rel_kernel(const float* __restrict__ rel,
                     unsigned short* __restrict__ relT,
                     const int* __restrict__ start_time_p)
{
    const int st = start_time_p[0];
    const long idx = (long)blockIdx.x * blockDim.x + threadIdx.x; // (x,y) pair
    const int x = (int)(idx >> 11);
    const int y = (int)(idx & (NXDIM - 1));
    const float* src = rel + ((long)x * NXDIM + y) * NTP1 + st;
    unsigned short* dst = relT + (long)x * NXDIM + y;
#pragma unroll
    for (int t = 0; t < NSTEPS; ++t) {
        dst[(long)t * NXDIM * NXDIM] = f2bf(src[t]);
    }
}

// ---------------------------------------------------------------------------
// z0 = factors[NT-1] : fp32 copy (for the identity-branch MLP math) and a
// transposed bf16 copy zT[z][y] used as the WMMA B operand.
// ---------------------------------------------------------------------------
__global__ void __launch_bounds__(256)
init_z_kernel(const float* __restrict__ factors,
              float* __restrict__ z_f32,
              unsigned short* __restrict__ zT)
{
    const int i = blockIdx.x * blockDim.x + threadIdx.x;  // = y*32 + z
    const int y = i >> 5;
    const int z = i & 31;
    const float v = factors[((23 * NXDIM) + y) * NZDIM + z];
    z_f32[i] = v;
    zT[z * NXDIM + y] = f2bf(v);
}

// ---------------------------------------------------------------------------
// Per-step GEMM:  part[ks][x][n] = sum_{y in k-slice ks} relT_t[x][y] * z[y][n]
// 128 threads = 4 waves; wave -> one 16-row M tile; block -> 64-row supertile
// and one of KSPLIT k-slices.  N = 32 -> two 16x16 accumulators per wave.
// ---------------------------------------------------------------------------
__global__ void __launch_bounds__(128)
gemm_step_kernel(const unsigned short* __restrict__ A_us,  // relT_t 2048x2048 bf16
                 const unsigned short* __restrict__ zT_us, // 32x2048 bf16
                 float* __restrict__ part)                 // KSPLIT x 2048 x 32
{
    const __bf16* A  = reinterpret_cast<const __bf16*>(A_us);
    const __bf16* zT = reinterpret_cast<const __bf16*>(zT_us);

    const int lane = threadIdx.x & 31;
    const int wave = threadIdx.x >> 5;
    const int ks   = blockIdx.x & (KSPLIT - 1);
    const int sm   = blockIdx.x >> 3;               // 0..31 (64-row supertile)
    const int m0   = sm * 64 + wave * 16;
    const int r16  = lane & 15;
    const int hi   = lane >> 4;                     // 0 or 1
    const int kbase = ks * KSLICE;

    const __bf16* aRow = A + (long)(m0 + r16) * NXDIM;
    const __bf16* bCol0 = zT + (long)r16 * NXDIM;          // columns 0..15
    const __bf16* bCol1 = zT + (long)(16 + r16) * NXDIM;   // columns 16..31

    v8f acc0 = {};
    v8f acc1 = {};

#pragma unroll
    for (int kc = 0; kc < KSLICE / 32; ++kc) {
        const int k0 = kbase + kc * 32;
        // A layout (matches B): lane<16 -> K {0..15}; lane>=16 -> {16..31}
        v8bf alo = *reinterpret_cast<const v8bf*>(aRow + k0 + hi * 16);
        v8bf ahi = *reinterpret_cast<const v8bf*>(aRow + k0 + hi * 16 + 8);
        v16bf a;
#pragma unroll
        for (int i = 0; i < 8; ++i) { a[i] = alo[i]; a[8 + i] = ahi[i]; }

        // B layout (matches A): lane holds 16 consecutive K of one column.
        const int kb = k0 + hi * 16;
        v16bf b0 = *reinterpret_cast<const v16bf*>(bCol0 + kb);
        v16bf b1 = *reinterpret_cast<const v16bf*>(bCol1 + kb);

        // prefetch next A chunk of this row (speculative; OOB is dropped)
        __builtin_prefetch(aRow + k0 + 32, 0, 1);

        acc0 = __builtin_amdgcn_wmma_f32_16x16x32_bf16(
                   false, a, false, b0, (short)0, acc0, false, false);
        acc1 = __builtin_amdgcn_wmma_f32_16x16x32_bf16(
                   false, a, false, b1, (short)0, acc1, false, false);
    }

    // C layout: VGPR v, lane<16 -> (M=v, N=lane); lane>=16 -> (M=8+v, N=lane-16)
    float* p = part + (long)ks * (NXDIM * NZDIM);
#pragma unroll
    for (int v = 0; v < 8; ++v) {
        const int x = m0 + v + hi * 8;
        p[x * NZDIM + r16]      = acc0[v];
        p[x * NZDIM + 16 + r16] = acc1[v];
    }
}

// ---------------------------------------------------------------------------
// Per-step finalize: reduce k-split partials -> inter, then
// z_next = tanh(z@W_dyn[0:32] + inter@W_dyn[32:64] + b)  (fp32),
// write z_gen, x_gen = z_next @ W_dec^T, and bf16 zT for the next step.
// 256 threads = 8 x-rows x 32 z-lanes.
// ---------------------------------------------------------------------------
__global__ void __launch_bounds__(256)
finalize_step_kernel(const float* __restrict__ part,
                     float* __restrict__ z_f32,
                     unsigned short* __restrict__ zT,
                     const float* __restrict__ W_dyn,   // 64 x 32
                     const float* __restrict__ b_dyn,   // 32
                     const float* __restrict__ W_dec,   // 8 x 32
                     float* __restrict__ out_x,         // NSTEPS x 2048 x 8
                     float* __restrict__ out_z,         // NSTEPS x 2048 x 32
                     int t)
{
    __shared__ float s_old[8][NZDIM];
    __shared__ float s_int[8][NZDIM];
    __shared__ float s_new[8][NZDIM];

    const int r  = threadIdx.x >> 5;
    const int zi = threadIdx.x & 31;
    const int x  = blockIdx.x * 8 + r;

    float inter = 0.f;
#pragma unroll
    for (int ks = 0; ks < KSPLIT; ++ks)
        inter += part[(long)ks * (NXDIM * NZDIM) + x * NZDIM + zi];

    s_int[r][zi] = inter;
    s_old[r][zi] = z_f32[x * NZDIM + zi];
    __syncthreads();

    float acc = b_dyn[zi];
#pragma unroll
    for (int j = 0; j < NZDIM; ++j) {
        acc += s_old[r][j] * W_dyn[j * NZDIM + zi];              // identity branch
        acc += s_int[r][j] * W_dyn[(NZDIM + j) * NZDIM + zi];    // relation branch
    }
    const float zn = tanhf(acc);

    z_f32[x * NZDIM + zi] = zn;
    zT[zi * NXDIM + x]    = f2bf(zn);
    out_z[(long)t * NXDIM * NZDIM + x * NZDIM + zi] = zn;
    s_new[r][zi] = zn;
    __syncthreads();

    if (zi < NDDIM) {
        float xg = 0.f;
#pragma unroll
        for (int j = 0; j < NZDIM; ++j)
            xg += s_new[r][j] * W_dec[zi * NZDIM + j];
        out_x[(long)t * NXDIM * NDDIM + x * NDDIM + zi] = xg;
    }
}

// ---------------------------------------------------------------------------
extern "C" void kernel_launch(void* const* d_in, const int* in_sizes, int n_in,
                              void* d_out, int out_size, void* d_ws, size_t ws_size,
                              hipStream_t stream)
{
    const float* relations = (const float*)d_in[0];
    const float* factors   = (const float*)d_in[1];
    const float* W_dyn     = (const float*)d_in[2];
    const float* b_dyn     = (const float*)d_in[3];
    const float* W_dec     = (const float*)d_in[4];
    const int*   start_t_p = (const int*)d_in[6];   // nsteps (d_in[5]) fixed = 16

    char* ws = (char*)d_ws;
    size_t off = 0;
    unsigned short* relT = (unsigned short*)(ws + off);
    off += (size_t)NSTEPS * NXDIM * NXDIM * sizeof(unsigned short);   // 134 MB
    unsigned short* zT = (unsigned short*)(ws + off);
    off += (size_t)NZDIM * NXDIM * sizeof(unsigned short);            // 128 KB
    float* z_f32 = (float*)(ws + off);
    off += (size_t)NXDIM * NZDIM * sizeof(float);                     // 256 KB
    float* part = (float*)(ws + off);
    off += (size_t)KSPLIT * NXDIM * NZDIM * sizeof(float);            // 2 MB

    float* out_x = (float*)d_out;                              // (16,2048,8)
    float* out_z = out_x + (size_t)NSTEPS * NXDIM * NDDIM;     // (16,2048,32)

    transpose_rel_kernel<<<(NXDIM * NXDIM) / 256, 256, 0, stream>>>(
        relations, relT, start_t_p);
    init_z_kernel<<<(NXDIM * NZDIM) / 256, 256, 0, stream>>>(
        factors, z_f32, zT);

    for (int t = 0; t < NSTEPS; ++t) {
        gemm_step_kernel<<<32 * KSPLIT, 128, 0, stream>>>(
            relT + (size_t)t * NXDIM * NXDIM, zT, part);
        finalize_step_kernel<<<NXDIM / 8, 256, 0, stream>>>(
            part, z_f32, zT, W_dyn, b_dyn, W_dec, out_x, out_z, t);
    }
}
